// Block_89361089561275
// MI455X (gfx1250) — compile-verified, matrix operand order checked by reading, not run on hardware
//
#include <hip/hip_runtime.h>
#include <hip/hip_bf16.h>

typedef __bf16 bf16_t;
typedef __attribute__((ext_vector_type(16))) __bf16 v16bf;
typedef __attribute__((ext_vector_type(8)))  __bf16 v8bf;
typedef __attribute__((ext_vector_type(8)))  float  v8f;

#define BB 8
#define NN 1024
#define CC 768
#define HH 12
#define DD 64
#define MLPH 3072
#define RED 256

// ---------------------------------------------------------------------------
// WMMA helpers
// ---------------------------------------------------------------------------
__device__ __forceinline__ v8f wmma_bf16(v16bf a, v16bf b, v8f c) {
    // D = A(16x32) * B(32x16) + C, f32 accumulate
    return __builtin_amdgcn_wmma_f32_16x16x32_bf16(
        /*neg_a=*/false, a, /*neg_b=*/false, b,
        /*c_mod=*/(short)0, c, /*reuse_a=*/false, /*reuse_b=*/false);
}

// Load a 16x32 bf16 fragment (A-layout; B uses the symmetric layout so the
// same addressing works when 'base' points at 16 rows of a row-major [*,K]
// tile).  Lane l: row = l&15, K-half = l>>4; elements are two contiguous
// 8-element (16B) chunks at K = kh*8 and K = 16 + kh*8.
__device__ __forceinline__ v16bf load_frag(const bf16_t* base, int ld, int lane) {
    const int row = lane & 15;
    const int kh  = lane >> 4;
    const bf16_t* p = base + row * ld + kh * 8;
    v8bf a0 = *(const v8bf*)(p);
    v8bf a1 = *(const v8bf*)(p + 16);
    v16bf f;
#pragma unroll
    for (int i = 0; i < 8; i++) { f[i] = a0[i]; f[i + 8] = a1[i]; }
    return f;
}

// ---------------------------------------------------------------------------
// CDNA5 async global->LDS copy (ASYNCcnt-tracked, bypasses VGPRs).
// Each lane moves 16 bytes: LDS[lds_off] = MEM[gptr].
// LDS offset = low 32 bits of the generic shared-pointer address
// ---------------------------------------------------------------------------
__device__ __forceinline__ void async_copy_b128(const bf16_t* lds_ptr,
                                                const bf16_t* gptr) {
    uint32_t lds_off = (uint32_t)(uintptr_t)lds_ptr;
    asm volatile("global_load_async_to_lds_b128 %0, %1, off"
                 :: "v"(lds_off), "v"(gptr)
                 : "memory");
}
__device__ __forceinline__ void wait_async0() {
    asm volatile("s_wait_asynccnt 0" ::: "memory");
}

// ---------------------------------------------------------------------------
// f32 -> bf16 convert
// ---------------------------------------------------------------------------
__global__ __launch_bounds__(256) void cvt_kernel(const float* __restrict__ in,
                                                  bf16_t* __restrict__ out, int n) {
    int i = blockIdx.x * 256 + threadIdx.x;
    if (i < n) out[i] = (bf16_t)in[i];
}

// ---------------------------------------------------------------------------
// LayerNorm over C=768, one row per block (256 threads, 3 elems each),
// bf16 output for the WMMA GEMMs.
// ---------------------------------------------------------------------------
__global__ __launch_bounds__(256) void ln_kernel(const float* __restrict__ x,
                                                 const float* __restrict__ g,
                                                 const float* __restrict__ be,
                                                 bf16_t* __restrict__ out) {
    const int row = blockIdx.x;
    const int tid = threadIdx.x;
    const float* xr = x + (size_t)row * CC;
    float v0 = xr[tid], v1 = xr[tid + 256], v2 = xr[tid + 512];
    float s = v0 + v1 + v2;
    float ss = v0 * v0 + v1 * v1 + v2 * v2;
#pragma unroll
    for (int off = 1; off < 32; off <<= 1) {
        s  += __shfl_xor(s, off, 32);
        ss += __shfl_xor(ss, off, 32);
    }
    __shared__ float sm[8], sq[8];
    __shared__ float mean_s, rstd_s;
    const int wave = tid >> 5, lane = tid & 31;
    if (lane == 0) { sm[wave] = s; sq[wave] = ss; }
    __syncthreads();
    if (tid == 0) {
        float S = 0.f, SS = 0.f;
#pragma unroll
        for (int i = 0; i < 8; i++) { S += sm[i]; SS += sq[i]; }
        float mean = S * (1.0f / CC);
        float var  = SS * (1.0f / CC) - mean * mean;
        mean_s = mean;
        rstd_s = rsqrtf(var + 1e-5f);
    }
    __syncthreads();
    const float mean = mean_s, rstd = rstd_s;
    bf16_t* orow = out + (size_t)row * CC;
    orow[tid]       = (bf16_t)((v0 - mean) * rstd * g[tid]       + be[tid]);
    orow[tid + 256] = (bf16_t)((v1 - mean) * rstd * g[tid + 256] + be[tid + 256]);
    orow[tid + 512] = (bf16_t)((v2 - mean) * rstd * g[tid + 512] + be[tid + 512]);
}

// ---------------------------------------------------------------------------
// Generic bf16 WMMA GEMM: C[M,Nt] = A[M,K] @ W[Nt,K]^T  (+ epilogue)
// Block: 256 threads = 8 waves, tile 128(M) x 64(N), K-step 32.
// Wave grid 4x2, each wave computes 32x32 (2x2 WMMA fragments).
// Double-buffered LDS filled by async global->LDS copies; pipeline:
//   wait(stage k) -> barrier -> issue(stage k+1, other buffer) -> wmma(k)
// ---------------------------------------------------------------------------
struct EpiParams {
    const float* bias;    // length Nt
    const float* bias2;   // v_bias (EPI 0)
    const float* gamma;   // gamma1 / gamma2
    const float* resid;   // x / x1 (f32)
    const float* temp;    // fc2 partial (f32)
    float*  outf;         // f32 output
    bf16_t* outbq;        // bf16 output (q / generic)
    bf16_t* outbk;        // bf16 k output
    bf16_t* outbv;        // bf16 v output
    float   scale;        // q scale
};

// EPI: 0=QKV split  1=bias+GeLU->bf16  2=bias+ReLU->bf16
//      3=proj+gamma1 residual -> f32   4=bias -> f32 temp
//      5=final: resid + gamma*(temp + acc + bias) -> f32
template <int EPI>
__global__ __launch_bounds__(256) void gemm_kernel(
    const bf16_t* __restrict__ A, int lda,
    const bf16_t* __restrict__ W, int ldw,
    int M, int Nt, int K, EpiParams P) {
    __shared__ alignas(16) bf16_t As[2][128 * 32];
    __shared__ alignas(16) bf16_t Bs[2][64 * 32];

    const int tid  = threadIdx.x;
    const int lane = tid & 31;
    const int wave = tid >> 5;
    const int wm = (wave >> 1) * 32;   // 0,32,64,96
    const int wn = (wave & 1) * 32;    // 0,32
    const int m0 = blockIdx.y * 128;
    const int n0 = blockIdx.x * 64;

    // per-thread 16B chunk coordinates for the tile copies
    const int crow = tid >> 2;          // 0..63
    const int ccol = (tid & 3) * 8;     // 0,8,16,24

    const v8f zero = {0.f, 0.f, 0.f, 0.f, 0.f, 0.f, 0.f, 0.f};
    v8f acc[2][2];
#pragma unroll
    for (int i = 0; i < 2; i++)
#pragma unroll
        for (int j = 0; j < 2; j++) acc[i][j] = zero;

    auto issue = [&](int k0, int buf) {
        // A tile 128x32: two 16B chunks per thread
        async_copy_b128(&As[buf][crow * 32 + ccol],
                        &A[(size_t)(m0 + crow) * lda + k0 + ccol]);
        async_copy_b128(&As[buf][(crow + 64) * 32 + ccol],
                        &A[(size_t)(m0 + crow + 64) * lda + k0 + ccol]);
        // B tile 64x32: one 16B chunk per thread
        async_copy_b128(&Bs[buf][crow * 32 + ccol],
                        &W[(size_t)(n0 + crow) * ldw + k0 + ccol]);
    };

    const int KT = K / 32;
    issue(0, 0);
    for (int kt = 0; kt < KT; kt++) {
        const int buf = kt & 1;
        wait_async0();          // my copies for stage kt are done
        __syncthreads();        // everyone's stage-kt tile is in LDS; prior
                                // readers of buf^1 have finished (iter kt-1)
        if (kt + 1 < KT) issue((kt + 1) * 32, buf ^ 1);
#pragma unroll
        for (int fm = 0; fm < 2; fm++) {
            v16bf a = load_frag(&As[buf][(wm + fm * 16) * 32], 32, lane);
#pragma unroll
            for (int fn = 0; fn < 2; fn++) {
                v16bf b = load_frag(&Bs[buf][(wn + fn * 16) * 32], 32, lane);
                acc[fm][fn] = wmma_bf16(a, b, acc[fm][fn]);
            }
        }
    }

    // Epilogue: C-layout element (j, lane) -> row j + 8*(lane>>4), col lane&15
    const int hlf = lane >> 4;
    const int nl  = lane & 15;
#pragma unroll
    for (int fm = 0; fm < 2; fm++)
#pragma unroll
        for (int fn = 0; fn < 2; fn++)
#pragma unroll
            for (int j = 0; j < 8; j++) {
                const int m = m0 + wm + fm * 16 + j + 8 * hlf;
                const int n = n0 + wn + fn * 16 + nl;
                const float v = acc[fm][fn][j];
                if constexpr (EPI == 0) {
                    const int b = m >> 10, s = m & 1023;
                    if (n < CC) {
                        float q = (v + P.bias[n]) * P.scale;
                        int h = n >> 6, d = n & 63;
                        P.outbq[(((size_t)(b * HH + h)) * NN + s) * DD + d] = (bf16_t)q;
                    } else if (n < 2 * CC) {
                        int c = n - CC;
                        int h = c >> 6, d = c & 63;
                        P.outbk[(((size_t)(b * HH + h)) * NN + s) * DD + d] = (bf16_t)v;
                    } else {
                        int c = n - 2 * CC;
                        float vv = v + P.bias2[c];
                        int h = c >> 6, d = c & 63;
                        P.outbv[(((size_t)(b * HH + h)) * NN + s) * DD + d] = (bf16_t)vv;
                    }
                } else if constexpr (EPI == 1) {
                    float t = v + P.bias[n];
                    float g = 0.5f * t * (1.0f + erff(t * 0.70710678118654752f));
                    P.outbq[(size_t)m * Nt + n] = (bf16_t)g;
                } else if constexpr (EPI == 2) {
                    float t = v + P.bias[n];
                    P.outbq[(size_t)m * Nt + n] = (bf16_t)fmaxf(t, 0.0f);
                } else if constexpr (EPI == 3) {
                    size_t o = (size_t)m * CC + n;
                    P.outf[o] = P.resid[o] + P.gamma[n] * (v + P.bias[n]);
                } else if constexpr (EPI == 4) {
                    P.outf[(size_t)m * CC + n] = v + P.bias[n];
                } else {  // EPI == 5
                    size_t o = (size_t)m * CC + n;
                    P.outf[o] = P.resid[o] + P.gamma[n] * (P.temp[o] + v + P.bias[n]);
                }
            }
}

// ---------------------------------------------------------------------------
// Flash attention.  Grid (N/128, H, B); block 256 = 8 waves; each wave owns a
// 16-row strip.  Q fragments live in registers; K tile async-copied to LDS;
// V transposed to LDS (B-layout); P goes C-layout -> LDS -> A-layout.
// Online softmax, rpb added per element, mask applied.  Output [B,N,C] bf16.
// ---------------------------------------------------------------------------
__global__ __launch_bounds__(256) void attn_kernel(
    const bf16_t* __restrict__ qg, const bf16_t* __restrict__ kg,
    const bf16_t* __restrict__ vg, const float* __restrict__ rpb,
    const unsigned char* __restrict__ mask, bf16_t* __restrict__ og) {
    __shared__ alignas(16) bf16_t Ks[64 * 64];        // [n][d]
    __shared__ alignas(16) bf16_t Vt[64 * 72];        // [d][k] padded
    __shared__ alignas(16) bf16_t Pb[8 * 16 * 72];    // per-wave P, padded

    const int tid = threadIdx.x, lane = tid & 31, wave = tid >> 5;
    const int b = blockIdx.z, h = blockIdx.y;
    const int r0 = blockIdx.x * 128 + wave * 16;
    const size_t bh = (size_t)(b * HH + h);
    const int hlf = lane >> 4, nl = lane & 15;

    // Q fragments (16 rows x 64 K) straight from global
    const bf16_t* qbase = qg + (bh * NN + r0) * DD;
    const v16bf qa0 = load_frag(qbase, DD, lane);
    const v16bf qa1 = load_frag(qbase + 32, DD, lane);

    const v8f zero = {0.f, 0.f, 0.f, 0.f, 0.f, 0.f, 0.f, 0.f};
    v8f oacc[4];
#pragma unroll
    for (int df = 0; df < 4; df++) oacc[df] = zero;
    float mi[8], li[8];
#pragma unroll
    for (int j = 0; j < 8; j++) { mi[j] = -1e30f; li[j] = 0.f; }

    bf16_t* Pw = &Pb[wave * 16 * 72];
    const int krow = tid >> 3, kcol = (tid & 7) * 8;   // K-tile chunk coords

    for (int ct = 0; ct < NN / 64; ct++) {
        const int c0 = ct * 64;
        __syncthreads();   // previous tile fully consumed
        {   // K tile [64 x 64]: async global->LDS, 2 x 16B per thread
            const bf16_t* kb = kg + (bh * NN + c0) * DD;
            async_copy_b128(&Ks[krow * 64 + kcol], &kb[krow * 64 + kcol]);
            async_copy_b128(&Ks[(krow + 32) * 64 + kcol],
                            &kb[(krow + 32) * 64 + kcol]);
            // V tile transposed (element scatter, VGPR path): Vt[d][k]=V[c0+k][d]
            const bf16_t* vb = vg + (bh * NN + c0) * DD;
            int kr = tid >> 2, d0 = (tid & 3) * 16;
            v8bf x0 = *(const v8bf*)&vb[kr * DD + d0];
            v8bf x1 = *(const v8bf*)&vb[kr * DD + d0 + 8];
#pragma unroll
            for (int i = 0; i < 8; i++) {
                Vt[(d0 + i) * 72 + kr]     = x0[i];
                Vt[(d0 + 8 + i) * 72 + kr] = x1[i];
            }
        }
        wait_async0();
        __syncthreads();

        // S = Q @ K^T  (16 x 64, four 16x16 fragments)
        float s[4][8];
#pragma unroll
        for (int fn = 0; fn < 4; fn++) {
            v8f a = zero;
            v16bf b0 = load_frag(&Ks[fn * 16 * 64], 64, lane);
            a = wmma_bf16(qa0, b0, a);
            v16bf b1 = load_frag(&Ks[fn * 16 * 64 + 32], 64, lane);
            a = wmma_bf16(qa1, b1, a);
#pragma unroll
            for (int j = 0; j < 8; j++) s[fn][j] = a[j];
        }

        // + rpb, mask
#pragma unroll
        for (int fn = 0; fn < 4; fn++) {
            const int cc = c0 + fn * 16 + nl;
            const bool ok = mask[b * NN + cc] != 0;
#pragma unroll
            for (int j = 0; j < 8; j++) {
                const int r = r0 + j + 8 * hlf;
                float sv = s[fn][j] + rpb[((size_t)h * NN + r) * NN + cc];
                s[fn][j] = ok ? sv : -1e30f;
            }
        }

        // online softmax (row reduction across 16 lanes of each half-wave)
#pragma unroll
        for (int j = 0; j < 8; j++) {
            float mx = fmaxf(fmaxf(s[0][j], s[1][j]), fmaxf(s[2][j], s[3][j]));
#pragma unroll
            for (int off = 1; off < 16; off <<= 1)
                mx = fmaxf(mx, __shfl_xor(mx, off, 32));
            const float mn = fmaxf(mi[j], mx);
            const float so = __expf(mi[j] - mn);
            float rs = 0.f;
#pragma unroll
            for (int fn = 0; fn < 4; fn++) {
                float p = __expf(s[fn][j] - mn);
                s[fn][j] = p;
                rs += p;
            }
#pragma unroll
            for (int off = 1; off < 16; off <<= 1)
                rs += __shfl_xor(rs, off, 32);
            li[j] = li[j] * so + rs;
            mi[j] = mn;
#pragma unroll
            for (int df = 0; df < 4; df++) oacc[df][j] *= so;
        }

        // P (C-layout) -> per-wave LDS (row-major) for A-layout reload
#pragma unroll
        for (int fn = 0; fn < 4; fn++)
#pragma unroll
            for (int j = 0; j < 8; j++)
                Pw[(j + 8 * hlf) * 72 + fn * 16 + nl] = (bf16_t)s[fn][j];

        // O += P @ V
#pragma unroll
        for (int kh = 0; kh < 2; kh++) {
            v16bf a = load_frag(Pw + kh * 32, 72, lane);
#pragma unroll
            for (int df = 0; df < 4; df++) {
                v16bf bb = load_frag(&Vt[df * 16 * 72 + kh * 32], 72, lane);
                oacc[df] = wmma_bf16(a, bb, oacc[df]);
            }
        }
    }

    // normalize + write [B,N,C] bf16  (feature = h*64 + df*16 + nl)
#pragma unroll
    for (int df = 0; df < 4; df++) {
#pragma unroll
        for (int j = 0; j < 8; j++) {
            const int r = r0 + j + 8 * hlf;
            const float ov = oacc[df][j] / li[j];
            og[((size_t)(b * NN + r)) * CC + h * DD + df * 16 + nl] = (bf16_t)ov;
        }
    }
}

// ---------------------------------------------------------------------------
// host launcher
// ---------------------------------------------------------------------------
extern "C" void kernel_launch(void* const* d_in, const int* in_sizes, int n_in,
                              void* d_out, int out_size, void* d_ws, size_t ws_size,
                              hipStream_t stream) {
    const float* x      = (const float*)d_in[0];
    const unsigned char* mask = (const unsigned char*)d_in[1];
    const float* rpb    = (const float*)d_in[2];
    const float* ln1_g  = (const float*)d_in[3];
    const float* ln1_b  = (const float*)d_in[4];
    const float* qkv_w  = (const float*)d_in[5];
    const float* q_bias = (const float*)d_in[6];
    const float* v_bias = (const float*)d_in[7];
    const float* proj_w = (const float*)d_in[8];
    const float* proj_b = (const float*)d_in[9];
    const float* gamma1 = (const float*)d_in[10];
    const float* ln2_g  = (const float*)d_in[11];
    const float* ln2_b  = (const float*)d_in[12];
    const float* fc1_w  = (const float*)d_in[13];
    const float* fc1_b  = (const float*)d_in[14];
    const float* fc2_w  = (const float*)d_in[15];
    const float* fc2_b  = (const float*)d_in[16];
    const float* ad_dw  = (const float*)d_in[17];
    const float* ad_db  = (const float*)d_in[18];
    const float* ad_uw  = (const float*)d_in[19];
    const float* ad_ub  = (const float*)d_in[20];
    const float* gamma2 = (const float*)d_in[21];

    const int M = BB * NN;  // 8192

    char* ws = (char*)d_ws;
    auto carve = [&](size_t bytes) -> char* {
        char* p = ws;
        ws += (bytes + 255) & ~(size_t)255;
        return p;
    };
    bf16_t* xn    = (bf16_t*)carve((size_t)M * CC * 2);
    bf16_t* qh    = (bf16_t*)carve((size_t)M * CC * 2);
    bf16_t* kh    = (bf16_t*)carve((size_t)M * CC * 2);
    bf16_t* vh    = (bf16_t*)carve((size_t)M * CC * 2);
    bf16_t* ob    = (bf16_t*)carve((size_t)M * CC * 2);
    float*  x1    = (float*)carve((size_t)M * CC * 4);
    bf16_t* xn2   = (bf16_t*)carve((size_t)M * CC * 2);
    bf16_t* hmid  = (bf16_t*)carve((size_t)M * MLPH * 2);
    bf16_t* aad   = (bf16_t*)carve((size_t)M * RED * 2);
    float*  tmpf  = (float*)carve((size_t)M * CC * 4);
    bf16_t* wqkv  = (bf16_t*)carve((size_t)3 * CC * CC * 2);
    bf16_t* wproj = (bf16_t*)carve((size_t)CC * CC * 2);
    bf16_t* wfc1  = (bf16_t*)carve((size_t)MLPH * CC * 2);
    bf16_t* wfc2  = (bf16_t*)carve((size_t)CC * MLPH * 2);
    bf16_t* wadd  = (bf16_t*)carve((size_t)RED * CC * 2);
    bf16_t* wadu  = (bf16_t*)carve((size_t)CC * RED * 2);

    auto cvt = [&](const float* src, bf16_t* dst, int n) {
        cvt_kernel<<<(n + 255) / 256, 256, 0, stream>>>(src, dst, n);
    };
    cvt(qkv_w, wqkv, 3 * CC * CC);
    cvt(proj_w, wproj, CC * CC);
    cvt(fc1_w, wfc1, MLPH * CC);
    cvt(fc2_w, wfc2, CC * MLPH);
    cvt(ad_dw, wadd, RED * CC);
    cvt(ad_uw, wadu, CC * RED);

    // LN1
    ln_kernel<<<M, 256, 0, stream>>>(x, ln1_g, ln1_b, xn);

    EpiParams P{};
    // QKV: [8192,768] @ [2304,768]^T, split into q(scaled)/k/v [B,H,N,D]
    P = EpiParams{};
    P.bias = q_bias; P.bias2 = v_bias;
    P.outbq = qh; P.outbk = kh; P.outbv = vh;
    P.scale = 0.125f;  // D^-0.5
    gemm_kernel<0><<<dim3(3 * CC / 64, M / 128), 256, 0, stream>>>(
        xn, CC, wqkv, CC, M, 3 * CC, CC, P);

    // attention
    attn_kernel<<<dim3(NN / 128, HH, BB), 256, 0, stream>>>(qh, kh, vh, rpb, mask, ob);

    // proj + gamma1 residual -> x1 (f32)
    P = EpiParams{};
    P.bias = proj_b; P.gamma = gamma1; P.resid = x; P.outf = x1;
    gemm_kernel<3><<<dim3(CC / 64, M / 128), 256, 0, stream>>>(
        ob, CC, wproj, CC, M, CC, CC, P);

    // LN2
    ln_kernel<<<M, 256, 0, stream>>>(x1, ln2_g, ln2_b, xn2);

    // fc1 + GeLU -> hmid bf16
    P = EpiParams{};
    P.bias = fc1_b; P.outbq = hmid;
    gemm_kernel<1><<<dim3(MLPH / 64, M / 128), 256, 0, stream>>>(
        xn2, CC, wfc1, CC, M, MLPH, CC, P);

    // adapter down + ReLU -> aad bf16
    P = EpiParams{};
    P.bias = ad_db; P.outbq = aad;
    gemm_kernel<2><<<dim3(RED / 64, M / 128), 256, 0, stream>>>(
        xn2, CC, wadd, CC, M, RED, CC, P);

    // fc2 -> tmp f32 (includes fc2_b)
    P = EpiParams{};
    P.bias = fc2_b; P.outf = tmpf;
    gemm_kernel<4><<<dim3(CC / 64, M / 128), 256, 0, stream>>>(
        hmid, MLPH, wfc2, MLPH, M, CC, MLPH, P);

    // adapter up + final: out = x1 + gamma2*(tmp + acc + ad_ub)
    P = EpiParams{};
    P.bias = ad_ub; P.gamma = gamma2; P.resid = x1; P.temp = tmpf;
    P.outf = (float*)d_out;
    gemm_kernel<5><<<dim3(CC / 64, M / 128), 256, 0, stream>>>(
        aad, RED, wadu, RED, M, CC, RED, P);
}
